// ComplexSpatialAttentionModule_33294586478991
// MI455X (gfx1250) — matrix operand order repaired and checked by reading, not compiled, not run on hardware
//
#include <hip/hip_runtime.h>
#include <stddef.h>

// ---------------------------------------------------------------------------
// Spatial self-attention (B=4, C=256, N=4096, dqk=32) for MI455X (gfx1250).
//
// Roofline: ~44 GFLOP total, dominated by N^2 attention terms. Mandatory HBM
// traffic ~45 MB (~2us @ 23.3 TB/s); a materialized attn matrix would add
// 536 MB (~23us) -> flash-attention formulation, everything through
// v_wmma_f32_16x16x32_bf16 (16384 FLOP/instr), f32 accumulate.
//
// CDNA5 specifics used:
//   * v_wmma_f32_16x16x32_bf16 for all GEMMs
//   * global_load_async_to_lds_b128 (ASYNCcnt) double-buffered K/V staging,
//     s_wait_asynccnt fences, workgroup barriers
//   * s_wait_dscnt fences for the intra-wave LDS softmax transpose
//   * v_cvt_pk_bf16_f32 packed conversion (guarded by __has_builtin)
// ---------------------------------------------------------------------------

static constexpr int Bc  = 4;
static constexpr int Cc  = 256;
static constexpr int CQK = 32;
static constexpr int Nn  = 4096;   // 64*64 tokens
static constexpr int KB  = Nn / 32;

typedef __attribute__((ext_vector_type(16))) __bf16        v16bf;
typedef __attribute__((ext_vector_type(8)))  float         v8f;
typedef __attribute__((ext_vector_type(4)))  unsigned int  uint4v;
typedef __attribute__((ext_vector_type(4)))  float         float4v;

union BFrag {
  v16bf          v;
  unsigned short us[16];
  unsigned       u[8];
  uint4v         u4[2];
};

static __device__ __forceinline__ unsigned short f2bf_us(float f) {
  unsigned u = __builtin_bit_cast(unsigned, f);
  unsigned r = u + 0x7FFFu + ((u >> 16) & 1u);   // round-to-nearest-even
  return (unsigned short)(r >> 16);
}

#if __has_builtin(__builtin_amdgcn_cvt_pk_bf16_f32)
static __device__ __forceinline__ unsigned cvt2(float a, float b) {
  typedef __attribute__((ext_vector_type(2))) __bf16 v2bf;
  v2bf r = __builtin_amdgcn_cvt_pk_bf16_f32(a, b);
  return __builtin_bit_cast(unsigned, r);
}
#else
static __device__ __forceinline__ unsigned cvt2(float a, float b) {
  return (unsigned)f2bf_us(a) | ((unsigned)f2bf_us(b) << 16);
}
#endif

static __device__ __forceinline__ v16bf load_bf16x16(const unsigned short* p) {
  BFrag f;
  f.u4[0] = *(const uint4v*)(p);
  f.u4[1] = *(const uint4v*)(p + 8);
  return f.v;
}

static __device__ __forceinline__ v8f wmma_bf16(v16bf a, v16bf b, v8f c) {
  // D = A(16x32) * B(32x16) + C, f32 accumulate
  return __builtin_amdgcn_wmma_f32_16x16x32_bf16(false, a, false, b,
                                                 (short)0, c, false, false);
}

// 16-byte async global->LDS copy (ASYNCcnt-tracked, gfx1250)
static __device__ __forceinline__ void async16(const unsigned short* lds_ptr,
                                               const unsigned short* gptr) {
  unsigned off = (unsigned)(size_t)lds_ptr;   // low 32 bits = LDS byte offset
  asm volatile("global_load_async_to_lds_b128 %0, %1, off"
               :: "v"(off), "v"(gptr) : "memory");
}

// ---------------------------------------------------------------------------
// Kernel 0: f32 -> bf16 conversion (x and the four weight matrices), so the
// GEMM inner loops carry no conversion VALU at all.
// ---------------------------------------------------------------------------
__global__ __launch_bounds__(256, 1)
void cvt_bf16_kernel(const float* __restrict__ s, unsigned short* __restrict__ d,
                     int n) {
  const int i = (blockIdx.x * 256 + threadIdx.x) * 4;
  if (i + 3 < n) {
    float4v f = *(const float4v*)(s + i);
    unsigned* dp = (unsigned*)(d + i);
    dp[0] = cvt2(f[0], f[1]);
    dp[1] = cvt2(f[2], f[3]);
  }
}

// ---------------------------------------------------------------------------
// Kernel 1: fused Q/K/V projections (all-bf16 operands).
//   Q,K -> token-major [B][N][32]; V -> [B][C][N] (contiguous attn fragments)
// One wave = one 16-token tile, K-loop over C in steps of 32.
// ---------------------------------------------------------------------------
__global__ __launch_bounds__(256, 1)
void qkv_proj_kernel(const unsigned short* __restrict__ xbf,
                     const unsigned short* __restrict__ wqb, const float* __restrict__ bq,
                     const unsigned short* __restrict__ wkb, const float* __restrict__ bk,
                     const unsigned short* __restrict__ wvb, const float* __restrict__ bv,
                     unsigned short* __restrict__ Qb,
                     unsigned short* __restrict__ Kb,
                     unsigned short* __restrict__ Vb) {
  const int lane = threadIdx.x & 31;
  const int wid  = threadIdx.x >> 5;
  const int gw   = blockIdx.x * 8 + wid;           // 0..1023
  const int b    = gw >> 8;
  const int n0   = (gw & 255) << 4;
  const int lm   = lane & 15;
  const int h    = lane >> 4;

  const unsigned short* xb = xbf + (size_t)b * Cc * Nn;

  const v8f z = {0.f,0.f,0.f,0.f,0.f,0.f,0.f,0.f};
  v8f qacc[2], kacc[2], vacc[16];
#pragma unroll
  for (int t = 0; t < 2; ++t) { qacc[t] = z; kacc[t] = z; }
#pragma unroll
  for (int t = 0; t < 16; ++t) vacc[t] = z;

  for (int c0 = 0; c0 < Cc; c0 += 32) {
    // A-fragment of x^T: M=token(lm), K=c (coalesced across the half-wave)
    BFrag ax;
#pragma unroll
    for (int j = 0; j < 8; ++j) {
      ax.us[j]     = xb[(size_t)(c0 + h * 16 +     j) * Nn + n0 + lm];
      ax.us[8 + j] = xb[(size_t)(c0 + h * 16 + 8 + j) * Nn + n0 + lm];
    }
    // B-fragment of x^T: K=c, N=token(lm)
    BFrag bx;
#pragma unroll
    for (int k = 0; k < 16; ++k)
      bx.us[k] = xb[(size_t)(c0 + h * 16 + k) * Nn + n0 + lm];

    // Q and K: D[m=token, n=out]; B = bf16 weight rows (straight b128 loads)
#pragma unroll
    for (int t = 0; t < 2; ++t) {
      v16bf bwq = load_bf16x16(wqb + (size_t)(t * 16 + lm) * Cc + c0 + h * 16);
      v16bf bwk = load_bf16x16(wkb + (size_t)(t * 16 + lm) * Cc + c0 + h * 16);
      qacc[t] = wmma_bf16(ax.v, bwq, qacc[t]);
      kacc[t] = wmma_bf16(ax.v, bwk, kacc[t]);
    }
    // V: D[m=c_out, n=token]; A = bf16 wv rows
#pragma unroll
    for (int t = 0; t < 16; ++t) {
      BFrag aw;
      const unsigned short* wvp = wvb + (size_t)(t * 16 + lm) * Cc + c0;
      aw.u4[0] = *(const uint4v*)(wvp + h * 16);
      aw.u4[1] = *(const uint4v*)(wvp + h * 16 + 8);
      vacc[t] = wmma_bf16(aw.v, bx.v, vacc[t]);
    }
  }

#pragma unroll
  for (int t = 0; t < 2; ++t) {
    const float bfq = bq[t * 16 + lm];
    const float bfk = bk[t * 16 + lm];
#pragma unroll
    for (int v = 0; v < 8; ++v) {
      const size_t row = (size_t)b * Nn + n0 + v + 8 * h;
      Qb[row * CQK + t * 16 + lm] = f2bf_us(qacc[t][v] + bfq);
      Kb[row * CQK + t * 16 + lm] = f2bf_us(kacc[t][v] + bfk);
    }
  }
#pragma unroll
  for (int t = 0; t < 16; ++t) {
#pragma unroll
    for (int v = 0; v < 8; ++v) {
      const int co = t * 16 + v + 8 * h;
      Vb[((size_t)b * Cc + co) * Nn + n0 + lm] = f2bf_us(vacc[t][v] + bv[co]);
    }
  }
}

// ---------------------------------------------------------------------------
// Kernel 2: flash attention, workgroup-cooperative.
// 8 waves/block (same batch) share each 32-key K (2KB) + V (16KB) slab,
// staged into LDS with double-buffered async global->LDS copies. One wave
// owns 16 query rows x 256 channels (128 accumulator VGPRs). Per key block:
// 2 WMMA (S = Q K^T) -> LDS transpose -> per-lane online softmax (state
// duplicated per lane-half, no shuffles) -> 16 WMMA (O += P V^T).
// LDS rows padded to 80B so the 16 fragment lanes hit disjoint bank quads.
// ---------------------------------------------------------------------------
__global__ __launch_bounds__(256, 1)
void attn_flash_kernel(const unsigned short* __restrict__ Qb,
                       const unsigned short* __restrict__ Kb,
                       const unsigned short* __restrict__ Vb,
                       unsigned short* __restrict__ AO) {
  __shared__ __attribute__((aligned(16))) unsigned short sK[2][32 * 40];
  __shared__ __attribute__((aligned(16))) unsigned short sV[2][256 * 40];
  __shared__ __attribute__((aligned(16))) float sSm[8][16 * 36 + 16];

  const int tid  = threadIdx.x;
  const int lane = tid & 31;
  const int wid  = tid >> 5;
  const int gw   = blockIdx.x * 8 + wid;
  const int b    = gw >> 8;                 // uniform per block (32 blk/batch)
  const int n0   = (gw & 255) << 4;
  const int lm   = lane & 15;
  const int h    = lane >> 4;

  float* sS = &sSm[wid][0];          // S tile [16][36]
  float* sA = &sSm[wid][16 * 36];    // per-row broadcast [16]

  // resident Q fragment (A layout: lane=M, K halves split by lane-half)
  BFrag qa;
  {
    const unsigned short* qp = Qb + ((size_t)b * Nn + n0 + lm) * CQK;
    qa.u4[0] = *(const uint4v*)(qp + h * 16);
    qa.u4[1] = *(const uint4v*)(qp + h * 16 + 8);
  }

  const v8f z = {0.f,0.f,0.f,0.f,0.f,0.f,0.f,0.f};
  v8f acc[16];
#pragma unroll
  for (int t = 0; t < 16; ++t) acc[t] = z;
  float Mrow = -1e30f, lrow = 0.0f;

  const int seg = tid & 3;         // 16B segment within a 64B row
  const int r4  = tid >> 2;

  // stage key-block 0 into buffer 0
  {
    if (tid < 128)
      async16(&sK[0][r4 * 40 + seg * 8],
              Kb + ((size_t)b * Nn + r4) * CQK + seg * 8);
#pragma unroll
    for (int i = 0; i < 4; ++i) {
      const int c = r4 + i * 64;
      async16(&sV[0][c * 40 + seg * 8],
              Vb + ((size_t)b * Cc + c) * Nn + seg * 8);
    }
  }

  for (int kb = 0; kb < KB; ++kb) {
    const int cur = kb & 1;
    asm volatile("s_wait_asynccnt 0x0" ::: "memory");
    __syncthreads();                         // buffer `cur` ready for all

    if (kb + 1 < KB) {                       // overlap: stage next block
      const int n2n = (kb + 1) * 32;
      if (tid < 128)
        async16(&sK[cur ^ 1][r4 * 40 + seg * 8],
                Kb + ((size_t)b * Nn + n2n + r4) * CQK + seg * 8);
#pragma unroll
      for (int i = 0; i < 4; ++i) {
        const int c = r4 + i * 64;
        async16(&sV[cur ^ 1][c * 40 + seg * 8],
                Vb + ((size_t)b * Cc + c) * Nn + n2n + seg * 8);
      }
    }

    // S = Q * K^T (K fragments straight from LDS, ds_load_b128 x2 each)
    v16bf kf0 = load_bf16x16(&sK[cur][lm * 40 + h * 16]);
    v16bf kf1 = load_bf16x16(&sK[cur][(16 + lm) * 40 + h * 16]);
    v8f s0 = wmma_bf16(qa.v, kf0, z);
    v8f s1 = wmma_bf16(qa.v, kf1, z);

    // D layout -> LDS: row m = v + 8h, col = key index
#pragma unroll
    for (int v = 0; v < 8; ++v) {
      sS[(v + 8 * h) * 36 + lm]      = s0[v];
      sS[(v + 8 * h) * 36 + 16 + lm] = s1[v];
    }
    asm volatile("s_wait_dscnt 0" ::: "memory");

    // each lane reads its full row (lanes L and L+16 duplicate row L&15)
    float r[32];
#pragma unroll
    for (int g = 0; g < 8; ++g) {
      float4v t4 = *(const float4v*)(sS + lm * 36 + g * 4);
      r[4*g+0] = t4[0]; r[4*g+1] = t4[1]; r[4*g+2] = t4[2]; r[4*g+3] = t4[3];
    }
    float mx = r[0];
#pragma unroll
    for (int i = 1; i < 32; ++i) mx = fmaxf(mx, r[i]);
    const float mnew  = fmaxf(Mrow, mx);
    const float alpha = __expf(Mrow - mnew);

    // exp + row-sum + pack P A-fragment (packed bf16 conversion)
    float eLo[8], eHi[8];
    float sum = 0.f;
#pragma unroll
    for (int j = 0; j < 8; ++j) {
      const float e0 = __expf(r[j]      - mnew);
      const float e1 = __expf(r[8 + j]  - mnew);
      const float e2 = __expf(r[16 + j] - mnew);
      const float e3 = __expf(r[24 + j] - mnew);
      sum += (e0 + e1) + (e2 + e3);
      eLo[j] = h ? e2 : e0;
      eHi[j] = h ? e3 : e1;
    }
    BFrag pf;
#pragma unroll
    for (int j = 0; j < 8; j += 2) {
      pf.u[j / 2]     = cvt2(eLo[j], eLo[j + 1]);
      pf.u[4 + j / 2] = cvt2(eHi[j], eHi[j + 1]);
    }
    lrow = lrow * alpha + sum;
    Mrow = mnew;

    // broadcast alpha per row through LDS, rescale C-layout accumulators
    if (lane < 16) sA[lane] = alpha;
    asm volatile("s_wait_dscnt 0" ::: "memory");
    float4v a0 = *(const float4v*)(sA + 8 * h);
    float4v a1 = *(const float4v*)(sA + 8 * h + 4);
    const float av[8] = {a0[0],a0[1],a0[2],a0[3],a1[0],a1[1],a1[2],a1[3]};
#pragma unroll
    for (int t = 0; t < 16; ++t)
#pragma unroll
      for (int v = 0; v < 8; ++v) acc[t][v] *= av[v];

    // O += P * V^T over all 16 channel tiles (V from LDS)
#pragma unroll
    for (int t = 0; t < 16; ++t) {
      v16bf vf = load_bf16x16(&sV[cur][(t * 16 + lm) * 40 + h * 16]);
      acc[t] = wmma_bf16(pf.v, vf, acc[t]);
    }
    __syncthreads();                 // all reads of `cur` done before refill
  }

  // final 1/l normalization (broadcast per row), store token-major bf16
  if (lane < 16) sA[lane] = 1.0f / lrow;
  asm volatile("s_wait_dscnt 0" ::: "memory");
  float4v l0 = *(const float4v*)(sA + 8 * h);
  float4v l1 = *(const float4v*)(sA + 8 * h + 4);
  const float lv[8] = {l0[0],l0[1],l0[2],l0[3],l1[0],l1[1],l1[2],l1[3]};
#pragma unroll
  for (int t = 0; t < 16; ++t)
#pragma unroll
    for (int v = 0; v < 8; ++v)
      AO[((size_t)b * Nn + n0 + v + 8 * h) * Cc + t * 16 + lm] =
          f2bf_us(acc[t][v] * lv[v]);
}

// ---------------------------------------------------------------------------
// Kernel 3: output projection + bias + residual (all-bf16 operands).
// D[m=o, n=token]; A = wo rows, B = AO token rows. Coalesced f32 stores.
// ---------------------------------------------------------------------------
__global__ __launch_bounds__(256, 1)
void out_proj_kernel(const float* __restrict__ x,
                     const unsigned short* __restrict__ wob,
                     const float* __restrict__ bo,
                     const unsigned short* __restrict__ AO,
                     float* __restrict__ out) {
  const int lane = threadIdx.x & 31;
  const int wid  = threadIdx.x >> 5;
  const int gw   = blockIdx.x * 8 + wid;
  const int b    = gw >> 8;
  const int n0   = (gw & 255) << 4;
  const int lm   = lane & 15;
  const int h    = lane >> 4;

  const v8f z = {0.f,0.f,0.f,0.f,0.f,0.f,0.f,0.f};
  v8f acc[16];
#pragma unroll
  for (int t = 0; t < 16; ++t) acc[t] = z;

  for (int c0 = 0; c0 < Cc; c0 += 32) {
    v16bf bao = load_bf16x16(AO + ((size_t)b * Nn + n0 + lm) * Cc + c0 + h * 16);
#pragma unroll
    for (int t = 0; t < 16; ++t) {
      BFrag aw;
      const unsigned short* wop = wob + (size_t)(t * 16 + lm) * Cc + c0;
      aw.u4[0] = *(const uint4v*)(wop + h * 16);
      aw.u4[1] = *(const uint4v*)(wop + h * 16 + 8);
      acc[t] = wmma_bf16(aw.v, bao, acc[t]);
    }
  }
#pragma unroll
  for (int t = 0; t < 16; ++t) {
#pragma unroll
    for (int v = 0; v < 8; ++v) {
      const int o = t * 16 + v + 8 * h;
      const size_t oi = ((size_t)b * Cc + o) * Nn + n0 + lm;
      out[oi] = acc[t][v] + bo[o] + x[oi];
    }
  }
}

// ---------------------------------------------------------------------------
extern "C" void kernel_launch(void* const* d_in, const int* in_sizes, int n_in,
                              void* d_out, int out_size, void* d_ws, size_t ws_size,
                              hipStream_t stream) {
  (void)in_sizes; (void)n_in; (void)out_size; (void)ws_size;
  const float* x  = (const float*)d_in[0];
  const float* wq = (const float*)d_in[1];
  const float* bq = (const float*)d_in[2];
  const float* wk = (const float*)d_in[3];
  const float* bk = (const float*)d_in[4];
  const float* wv = (const float*)d_in[5];
  const float* bv = (const float*)d_in[6];
  const float* wo = (const float*)d_in[7];
  const float* bo = (const float*)d_in[8];
  float* out = (float*)d_out;

  // Workspace (bf16 ushorts):
  //   Q[4][4096][32] | K[4][4096][32] | V[4][256][4096] | AO[4][4096][256]
  //   | xbf[4][256][4096] | wq/wk[32][256] | wv/wo[256][256]   (~27.9 MB)
  unsigned short* Qb  = (unsigned short*)d_ws;
  unsigned short* Kb  = Qb  + (size_t)Bc * Nn * CQK;
  unsigned short* Vb  = Kb  + (size_t)Bc * Nn * CQK;
  unsigned short* AO  = Vb  + (size_t)Bc * Cc * Nn;
  unsigned short* xbf = AO  + (size_t)Bc * Nn * Cc;
  unsigned short* wqb = xbf + (size_t)Bc * Cc * Nn;
  unsigned short* wkb = wqb + (size_t)CQK * Cc;
  unsigned short* wvb = wkb + (size_t)CQK * Cc;
  unsigned short* wob = wvb + (size_t)Cc * Cc;

  dim3 block(256);
  const int nx = Bc * Cc * Nn;     // 4194304
  const int nqk = CQK * Cc;        // 8192
  const int ncc = Cc * Cc;         // 65536
  cvt_bf16_kernel<<<nx / 1024, block, 0, stream>>>(x,  xbf, nx);
  cvt_bf16_kernel<<<nqk / 1024, block, 0, stream>>>(wq, wqb, nqk);
  cvt_bf16_kernel<<<nqk / 1024, block, 0, stream>>>(wk, wkb, nqk);
  cvt_bf16_kernel<<<ncc / 1024, block, 0, stream>>>(wv, wvb, ncc);
  cvt_bf16_kernel<<<ncc / 1024, block, 0, stream>>>(wo, wob, ncc);

  dim3 grid(Bc * (Nn / 16) / 8);   // 1024 waves, 8 waves (256 thr) per block
  qkv_proj_kernel<<<grid, block, 0, stream>>>(xbf, wqb, bq, wkb, bk, wvb, bv,
                                              Qb, Kb, Vb);
  attn_flash_kernel<<<grid, block, 0, stream>>>(Qb, Kb, Vb, AO);
  out_proj_kernel<<<grid, block, 0, stream>>>(x, wob, bo, AO, out);
}
